// HeterogeneousEdgeUpdate_50173807952923
// MI455X (gfx1250) — compile-verified, matrix operand order checked by reading, not run on hardware
//
#include <hip/hip_runtime.h>

#define TT 4
#define CC 16
#define BBASIS 32
#define NNODE 8192
#define EDG 32768
#define DD 512
#define DOUT 1024

typedef __attribute__((ext_vector_type(16))) __bf16 v16bf;
typedef __attribute__((ext_vector_type(8)))  float v8f;

union FragU { v16bf v; uint4 q[2]; };

__device__ __forceinline__ unsigned short f2bf(float f) {
    unsigned u = __float_as_uint(f);
    unsigned r = u + 0x7FFFu + ((u >> 16) & 1u);
    return (unsigned short)(r >> 16);
}
__device__ __forceinline__ float bf2f(unsigned short h) {
    return __uint_as_float(((unsigned)h) << 16);
}

// ---------------- workspace layout (bytes) ----------------
#define OFF_COUNTS 0                         // TT ints
#define OFF_BUCKET 256                       // TT*EDG ints       = 512 KB
#define OFF_W1BF   (OFF_BUCKET + TT*EDG*4)   // TT*DD*DD bf16     = 2 MB
#define OFF_W2BF   (OFF_W1BF + TT*DD*DD*2)   // TT*DOUT*DD bf16   = 4 MB
#define OFF_FEAT   (OFF_W2BF + TT*DOUT*DD*2) // EDG*DD bf16       = 32 MB
// total ~ 40.4 MB

// ---------------- kernel 0: init counters ----------------
__global__ void k_init(int* counts) {
    if (threadIdx.x < TT) counts[threadIdx.x] = 0;
}

// ---------------- kernel 1: transpose + bf16 convert weights ----------------
// src: [T][K][Nc] f32  ->  dst: [T][Nc][K] bf16
__global__ void k_convert_w(const float* __restrict__ src,
                            unsigned short* __restrict__ dst,
                            int K, int Nc) {
    __shared__ float tile[32][33];
    int t  = blockIdx.z;
    int k0 = blockIdx.y * 32;
    int n0 = blockIdx.x * 32;
    int tx = threadIdx.x, ty = threadIdx.y;
#pragma unroll
    for (int i = 0; i < 4; ++i) {
        int k = ty + i * 8;
        tile[k][tx] = src[((size_t)t * K + (k0 + k)) * Nc + n0 + tx];
    }
    __syncthreads();
#pragma unroll
    for (int i = 0; i < 4; ++i) {
        int n = ty + i * 8;
        dst[((size_t)t * Nc + (n0 + n)) * K + k0 + tx] = f2bf(tile[tx][n]);
    }
}

// ---------------- kernel 2: bucket edges by source-node type ----------------
__global__ void k_bucket(const int* __restrict__ ei, const int* __restrict__ an,
                         int* counts, int* bucket) {
    int e = blockIdx.x * 256 + threadIdx.x;
    if (e < EDG) {
        int t = an[ei[e]];
        int pos = atomicAdd(&counts[t], 1);
        bucket[t * EDG + pos] = e;
    }
}

// ---------------- kernel 3: bilinear overlaps -> feat (bf16) ----------------
// feat[e] = [ (A^T M Bf).flatten , (Xa^T M Xb).flatten ]  (512 values)
__global__ void k_prep(const float* __restrict__ x, const int* __restrict__ ei,
                       const float* __restrict__ efa, const float* __restrict__ efb,
                       const float* __restrict__ em,
                       unsigned short* __restrict__ featbf) {
    __shared__ float sM[32 * 32];
    __shared__ float sBf[32 * 16], sXb[32 * 16], sA[32 * 16], sXa[32 * 16];
    __shared__ float t1e[32 * 16], t1n[32 * 16];
    int e = blockIdx.x, tid = threadIdx.x;
    int src = ei[e], dst = ei[EDG + e];

    const float4* Mg = (const float4*)(em + (size_t)e * 1024);
    ((float4*)sM)[tid]        = Mg[tid];
    ((float4*)sM)[tid + 128]  = Mg[tid + 128];
    ((float4*)sA)[tid]  = ((const float4*)(efa + (size_t)e * 512))[tid];
    ((float4*)sBf)[tid] = ((const float4*)(efb + (size_t)e * 512))[tid];
    ((float4*)sXa)[tid] = ((const float4*)(x + (size_t)src * 512))[tid];
    ((float4*)sXb)[tid] = ((const float4*)(x + (size_t)dst * 512))[tid];
    __syncthreads();

    // t1 = M @ {Bf, Xb}  (32x16 each)
    for (int idx = tid; idx < 512; idx += 128) {
        int i = idx >> 4, m = idx & 15;
        float se = 0.f, sn = 0.f;
#pragma unroll
        for (int j = 0; j < 32; ++j) {
            float mm = sM[i * 32 + j];
            se += mm * sBf[j * 16 + m];
            sn += mm * sXb[j * 16 + m];
        }
        t1e[idx] = se; t1n[idx] = sn;
    }
    __syncthreads();

    unsigned short* fout = featbf + (size_t)e * DD;
    for (int idx = tid; idx < 256; idx += 128) {
        int c = idx >> 4, m = idx & 15;
        float se = 0.f, sn = 0.f;
#pragma unroll
        for (int i = 0; i < 32; ++i) {
            se += sA[i * 16 + c]  * t1e[i * 16 + m];
            sn += sXa[i * 16 + c] * t1n[i * 16 + m];
        }
        fout[idx]       = f2bf(se);
        fout[256 + idx] = f2bf(sn);
    }
}

// ---------------- kernel 4: fused per-type MLP (WMMA) + apply ----------------
// block = 256 threads = 8 waves; tile = 16 same-type edges
__global__ void __launch_bounds__(256)
k_mlp(const float* __restrict__ x, const int* __restrict__ ei,
      const float* __restrict__ efa, const float* __restrict__ efb,
      const int* __restrict__ counts, const int* __restrict__ bucket,
      const unsigned short* __restrict__ w1bf, const unsigned short* __restrict__ w2bf,
      const unsigned short* __restrict__ featbf,
      const float* __restrict__ b1, const float* __restrict__ lng,
      const float* __restrict__ lnb, const float* __restrict__ b2,
      float* __restrict__ out) {
    __shared__ __align__(16) unsigned short s_A[16 * DD];    // feat, later h_ln (bf16)
    __shared__ __align__(16) unsigned short s_W[16 * DOUT];  // normalized weights (bf16)
    __shared__ int s_eid[16];
    __shared__ int s_sd[32];                                 // [0..15]=src, [16..31]=dst
    __shared__ float s_sum[16], s_ssq[16], s_mu[16], s_rs[16];
    __shared__ float s_q[16 * 4];

    int tid = threadIdx.x;
    // --- map block -> (type, tile) ---
    int bid = blockIdx.x;
    int t = -1, tile = 0, acc = 0;
    for (int q = 0; q < TT; ++q) {
        int c = counts[q];
        int tl = (c + 15) >> 4;
        if (t < 0 && bid < acc + tl) { t = q; tile = bid - acc; }
        acc += tl;
    }
    if (t < 0) return;
    int cbase = tile * 16;
    int nval = counts[t] - cbase; if (nval > 16) nval = 16;

    if (tid < 16) {
        int e = bucket[t * EDG + cbase + ((tid < nval) ? tid : 0)];
        s_eid[tid] = e;
        s_sd[tid] = ei[e];
        s_sd[tid + 16] = ei[EDG + e];
    }
    if (tid < 16) { s_sum[tid] = 0.f; s_ssq[tid] = 0.f; }
    if (tid < 64) s_q[tid] = 0.f;
    __syncthreads();

    // --- stage feat tile into LDS (1024 x 16B chunks) ---
#pragma unroll
    for (int i = 0; i < 4; ++i) {
        int chunk = tid * 4 + i;
        int row = chunk >> 6;
        ((uint4*)s_A)[chunk] =
            ((const uint4*)(featbf + (size_t)s_eid[row] * DD))[chunk & 63];
    }
    __syncthreads();

    int wv = tid >> 5, lane = tid & 31;
    int half = lane >> 4, l15 = lane & 15;

    // hoisted per-lane base pointers: all fragment addresses become
    // base + compile-time immediate offsets (j*16*DD and k0 within rows)
    const unsigned short* ap  = s_A + l15 * DD + half * 8;
    const unsigned short* wp1 =
        w1bf + ((size_t)(t * DD + wv * 64 + l15)) * DD + half * 8;
    const unsigned short* wp2 =
        w2bf + ((size_t)(t * DOUT + wv * 128 + l15)) * DD + half * 8;

    v8f zero = {0.f, 0.f, 0.f, 0.f, 0.f, 0.f, 0.f, 0.f};

    // ================= GEMM1: h = feat @ W1[t]  (16x512) =================
    v8f acc1[4];
#pragma unroll
    for (int j = 0; j < 4; ++j) acc1[j] = zero;

#pragma unroll
    for (int k0 = 0; k0 < DD; k0 += 32) {
        FragU fa;
        fa.q[0] = *(const uint4*)(ap + k0);
        fa.q[1] = *(const uint4*)(ap + k0 + 16);
        FragU fb[4];
#pragma unroll
        for (int j = 0; j < 4; ++j) {   // batch all B loads first
            fb[j].q[0] = *(const uint4*)(wp1 + j * 16 * DD + k0);
            fb[j].q[1] = *(const uint4*)(wp1 + j * 16 * DD + k0 + 16);
        }
#pragma unroll
        for (int j = 0; j < 4; ++j)
            acc1[j] = __builtin_amdgcn_wmma_f32_16x16x32_bf16(
                false, fa.v, false, fb[j].v, (short)0, acc1[j], false, false);
    }

    // --- epilogue1: bias + SiLU, row stats for LayerNorm ---
    float hval[4][8];
    float gj[4], bj[4];
#pragma unroll
    for (int j = 0; j < 4; ++j) {
        int n = (wv * 4 + j) * 16 + l15;
        float bias = b1[t * DD + n];
        gj[j] = lng[t * DD + n];
        bj[j] = lnb[t * DD + n];
#pragma unroll
        for (int r = 0; r < 8; ++r) {
            float v = acc1[j][r] + bias;
            hval[j][r] = v / (1.f + __expf(-v));
        }
    }
#pragma unroll
    for (int r = 0; r < 8; ++r) {
        float ps = 0.f, pq = 0.f;
#pragma unroll
        for (int j = 0; j < 4; ++j) { ps += hval[j][r]; pq += hval[j][r] * hval[j][r]; }
        atomicAdd(&s_sum[r + half * 8], ps);
        atomicAdd(&s_ssq[r + half * 8], pq);
    }
    __syncthreads();
    if (tid < 16) {
        float mu = s_sum[tid] * (1.f / DD);
        float var = s_ssq[tid] * (1.f / DD) - mu * mu;
        s_mu[tid] = mu;
        s_rs[tid] = rsqrtf(var + 1e-5f);
    }
    __syncthreads();

    // --- LayerNorm; write h_ln (bf16) over s_A ---
#pragma unroll
    for (int j = 0; j < 4; ++j) {
        int n = (wv * 4 + j) * 16 + l15;
#pragma unroll
        for (int r = 0; r < 8; ++r) {
            int row = r + half * 8;
            float hn = (hval[j][r] - s_mu[row]) * s_rs[row] * gj[j] + bj[j];
            s_A[row * DD + n] = f2bf(hn);
        }
    }
    __syncthreads();

    // ================= GEMM2: w = h_ln @ W2[t]  (16x1024) =================
    v8f acc2[8];
#pragma unroll
    for (int j = 0; j < 8; ++j) acc2[j] = zero;

#pragma unroll
    for (int k0 = 0; k0 < DD; k0 += 32) {
        FragU fa;
        fa.q[0] = *(const uint4*)(ap + k0);
        fa.q[1] = *(const uint4*)(ap + k0 + 16);
        FragU fb[8];
#pragma unroll
        for (int j = 0; j < 8; ++j) {   // batch all B loads first
            fb[j].q[0] = *(const uint4*)(wp2 + j * 16 * DD + k0);
            fb[j].q[1] = *(const uint4*)(wp2 + j * 16 * DD + k0 + 16);
        }
#pragma unroll
        for (int j = 0; j < 8; ++j)
            acc2[j] = __builtin_amdgcn_wmma_f32_16x16x32_bf16(
                false, fa.v, false, fb[j].v, (short)0, acc2[j], false, false);
    }

    // --- epilogue2: bias, Frobenius-norm per (edge,segment) ---
    int seg = wv >> 1;  // this wave's N range lies in one 256-col segment
    float wval[8][8];
#pragma unroll
    for (int j = 0; j < 8; ++j) {
        int n = (wv * 8 + j) * 16 + l15;
        float bias = b2[t * DOUT + n];
#pragma unroll
        for (int r = 0; r < 8; ++r) wval[j][r] = acc2[j][r] + bias;
    }
#pragma unroll
    for (int r = 0; r < 8; ++r) {
        float pq = 0.f;
#pragma unroll
        for (int j = 0; j < 8; ++j) pq += wval[j][r] * wval[j][r];
        atomicAdd(&s_q[(r + half * 8) * 4 + seg], pq);
    }
    __syncthreads();
    if (tid < 64) s_q[tid] = 1.f / (sqrtf(s_q[tid]) + 1e-6f);
    __syncthreads();
#pragma unroll
    for (int j = 0; j < 8; ++j) {
        int n = (wv * 8 + j) * 16 + l15;
#pragma unroll
        for (int r = 0; r < 8; ++r) {
            int row = r + half * 8;
            s_W[row * DOUT + n] = f2bf(wval[j][r] * s_q[row * 4 + seg]);
        }
    }
    __syncthreads();

    // ================= apply: out = 0.5*(EF @ We + XF @ Wn) =================
#pragma unroll
    for (int it = 0; it < 4; ++it) {
        int task = tid + it * 256;       // 1024 tasks: (edge r)x(a/b)x(brow)
        int brow = task & 31;
        int ab = (task >> 5) & 1;
        int r = task >> 6;
        int e = s_eid[r];
        const float* EF = (ab ? efb : efa) + ((size_t)e * BBASIS + brow) * CC;
        const float* XF = x + ((size_t)s_sd[ab * 16 + r] * BBASIS + brow) * CC;
        const unsigned short* We = &s_W[r * DOUT + ab * 256];
        const unsigned short* Wn = &s_W[r * DOUT + (2 + ab) * 256];
        float o[16];
#pragma unroll
        for (int q = 0; q < 16; ++q) o[q] = 0.f;
#pragma unroll
        for (int p = 0; p < 16; ++p) {
            float ev = EF[p], xv = XF[p];
#pragma unroll
            for (int q = 0; q < 16; ++q)
                o[q] += ev * bf2f(We[p * 16 + q]) + xv * bf2f(Wn[p * 16 + q]);
        }
        if (r < nval) {
            float* OUT = out + (size_t)ab * EDG * BBASIS * CC
                             + ((size_t)e * BBASIS + brow) * CC;
#pragma unroll
            for (int q = 0; q < 16; ++q) OUT[q] = 0.5f * o[q];
        }
    }
}

// ---------------- launch ----------------
extern "C" void kernel_launch(void* const* d_in, const int* in_sizes, int n_in,
                              void* d_out, int out_size, void* d_ws, size_t ws_size,
                              hipStream_t stream) {
    const float* x   = (const float*)d_in[0];
    const int*   ei  = (const int*)d_in[1];
    const float* efa = (const float*)d_in[2];
    const float* efb = (const float*)d_in[3];
    const float* em  = (const float*)d_in[4];
    const int*   an  = (const int*)d_in[5];
    const float* W1  = (const float*)d_in[6];
    const float* b1  = (const float*)d_in[7];
    const float* lng = (const float*)d_in[8];
    const float* lnb = (const float*)d_in[9];
    const float* W2  = (const float*)d_in[10];
    const float* b2  = (const float*)d_in[11];
    float* out = (float*)d_out;

    char* ws = (char*)d_ws;   // needs ~40.4 MB
    int* counts  = (int*)(ws + OFF_COUNTS);
    int* bucket  = (int*)(ws + OFF_BUCKET);
    unsigned short* w1bf = (unsigned short*)(ws + OFF_W1BF);
    unsigned short* w2bf = (unsigned short*)(ws + OFF_W2BF);
    unsigned short* featbf = (unsigned short*)(ws + OFF_FEAT);

    k_init<<<1, 32, 0, stream>>>(counts);
    k_convert_w<<<dim3(DD / 32, DD / 32, TT), dim3(32, 8), 0, stream>>>(W1, w1bf, DD, DD);
    k_convert_w<<<dim3(DOUT / 32, DD / 32, TT), dim3(32, 8), 0, stream>>>(W2, w2bf, DD, DOUT);
    k_bucket<<<EDG / 256, 256, 0, stream>>>(ei, an, counts, bucket);
    k_prep<<<EDG, 128, 0, stream>>>(x, ei, efa, efb, em, featbf);
    k_mlp<<<EDG / 16 + TT, 256, 0, stream>>>(x, ei, efa, efb, counts, bucket,
                                             w1bf, w2bf, featbf,
                                             b1, lng, lnb, b2, out);
}
